// GPT_56659208569190
// MI455X (gfx1250) — matrix operand order repaired and checked by reading, not compiled, not run on hardware
//
#include <hip/hip_runtime.h>
#include <hip/hip_bf16.h>
#include <math.h>
#include <stdint.h>

// ---------------------------------------------------------------------------
// GPT forward for MI455X (gfx1250, wave32, WMMA bf16 16x16x32).
// All GEMMs and attention matmuls run as v_wmma_f32_16x16x32_bf16 (f32 acc).
// Fragment groups are preloaded so loads clause together and WMMAs overlap
// outstanding loads via staggered s_wait_loadcnt (instead of full drains).
// ---------------------------------------------------------------------------

typedef __bf16 bf16_t;
typedef __attribute__((ext_vector_type(16))) __bf16 v16bf;
typedef __attribute__((ext_vector_type(8)))  float  v8f;

#define T_SEQ  2048
#define VOCAB  32768
#define NLAY   12
#define NHEAD  6
#define NKVH   2
#define CDIM   768
#define HDIM   128
#define KVDIM  256
#define WINSZ  512
#define SOFTCAP_ 15.0f

union FragAB { v16bf v; uint32_t u[8]; unsigned short h[16]; };
union FragC  { v8f   v; float f[8]; };

// ISA 7.12.2: 16-bit A 16x32 — lanes 0-15 hold K {0..7,16..23}, lanes 16-31 +8.
__device__ __forceinline__ int koffA(int j, int half) {
    return (j < 4 ? 2 * j : 2 * j + 8) + half * 8;
}
// 16-bit B 32x16 — same K order as A: lanes 0-15 hold K {0..7,16..23}, lanes 16-31 +8.
__device__ __forceinline__ int koffB(int j, int half) {
    return (j < 4 ? 2 * j : 2 * j + 8) + half * 8;
}

__device__ __forceinline__ v8f wmma_bf16(v16bf a, v16bf b, v8f c) {
    return __builtin_amdgcn_wmma_f32_16x16x32_bf16(
        /*neg_a=*/false, a, /*neg_b=*/false, b,
        /*c_mod=*/(short)0, c, /*reuse_a=*/false, /*reuse_b=*/false);
}

// ---------------------------------------------------------------------------
// block-wide sum (wave32 shuffles + LDS across waves)
// ---------------------------------------------------------------------------
__device__ float block_sum(float v, float* sb) {
    for (int m = 16; m >= 1; m >>= 1) v += __shfl_xor(v, m, 32);
    int w = threadIdx.x >> 5, ln = threadIdx.x & 31;
    if (ln == 0) sb[w] = v;
    __syncthreads();
    float r = (threadIdx.x < (blockDim.x >> 5)) ? sb[threadIdx.x] : 0.f;
    if (w == 0) {
        for (int m = 16; m >= 1; m >>= 1) r += __shfl_xor(r, m, 32);
        if (ln == 0) sb[0] = r;
    }
    __syncthreads();
    float out = sb[0];
    __syncthreads();
    return out;
}

// ---------------------------------------------------------------------------
// fp32 -> bf16 weight conversion
// ---------------------------------------------------------------------------
__global__ void k_cvt(const float* __restrict__ s, bf16_t* __restrict__ d, long n) {
    long i = (long)blockIdx.x * 256 + threadIdx.x;
    if (i < n) d[i] = (bf16_t)s[i];
}

// ---------------------------------------------------------------------------
// Embedding: x = rms(wte[idx]); x += smear_lambda*sigmoid(x[:,:24]@sgw)*x_prev
// one block (256 threads, 3 channels each) per token
// ---------------------------------------------------------------------------
__global__ void __launch_bounds__(256) k_embed(
    const int* __restrict__ idx, const float* __restrict__ wte,
    const float* __restrict__ sgw, const float* __restrict__ slam,
    float* __restrict__ x, float* __restrict__ x0)
{
    __shared__ float sb[16];
    int t   = blockIdx.x;
    int id  = idx[t];
    int idp = idx[t > 0 ? t - 1 : 0];
    float e[3], ep[3];
    for (int j = 0; j < 3; j++) {
        int c = threadIdx.x + j * 256;
        e[j]  = wte[(size_t)id  * CDIM + c];
        ep[j] = wte[(size_t)idp * CDIM + c];
    }
    float ss = e[0]*e[0] + e[1]*e[1] + e[2]*e[2];
    ss = block_sum(ss, sb);
    float r1 = rsqrtf(ss / CDIM + 1e-6f);
    float ssp = ep[0]*ep[0] + ep[1]*ep[1] + ep[2]*ep[2];
    ssp = block_sum(ssp, sb);
    float r2 = rsqrtf(ssp / CDIM + 1e-6f);
    // smear gate over first 24 normed channels (channel c==tid for j==0)
    float part = (threadIdx.x < 24) ? (e[0] * r1) * sgw[threadIdx.x] : 0.f;
    float z = block_sum(part, sb);
    float s = slam[0] / (1.f + __expf(-z));
    for (int j = 0; j < 3; j++) {
        int c = threadIdx.x + j * 256;
        float vv = e[j] * r1 + s * (ep[j] * r2);
        x [(size_t)t * CDIM + c] = vv;
        x0[(size_t)t * CDIM + c] = vv;
    }
}

// ---------------------------------------------------------------------------
// x = a*x + b*other  (a,b from device pointers, optional), then h = rms(x)
// writes h as f32 (for ve-gates) and bf16 (for WMMA GEMMs)
// ---------------------------------------------------------------------------
__global__ void __launch_bounds__(256) k_norm(
    float* __restrict__ x, const float* __restrict__ other,
    const float* pa, int ai, const float* pb, int bi, float bsign,
    float* __restrict__ hf, bf16_t* __restrict__ hb)
{
    __shared__ float sb[16];
    int t = blockIdx.x;
    float a = pa ? pa[ai] : 1.f;
    float b = pb ? pb[bi] * bsign : 0.f;
    float v[3];
    for (int j = 0; j < 3; j++) {
        int c = threadIdx.x + j * 256;
        float xv = a * x[(size_t)t * CDIM + c] + b * other[(size_t)t * CDIM + c];
        x[(size_t)t * CDIM + c] = xv;
        v[j] = xv;
    }
    float ss = v[0]*v[0] + v[1]*v[1] + v[2]*v[2];
    ss = block_sum(ss, sb);
    float r = rsqrtf(ss / CDIM + 1e-6f);
    for (int j = 0; j < 3; j++) {
        int c = threadIdx.x + j * 256;
        float hv = v[j] * r;
        hf[(size_t)t * CDIM + c] = hv;
        hb[(size_t)t * CDIM + c] = (bf16_t)hv;
    }
}

// ---------------------------------------------------------------------------
// WMMA GEMM: out[M,N] = A[M,K] @ W[N,K]^T, bf16 in, f32 accumulate.
// 256 threads = 8 waves (4 in M x 2 in N); block tile 64x128.
// Each wave computes a 16x64 strip. Per K-step the A fragment and all four
// B fragments are preloaded (one clause of b128 loads), then 4 WMMAs issue
// against staggered load counts.
// Epilogues: 0=store f32, 1=resid+acc -> f32 (in place), 2=relu(acc)^2 -> bf16,
//            3=softcap tanh -> f32
// ---------------------------------------------------------------------------
template <int MODE>
__global__ void __launch_bounds__(256) k_gemm(
    const bf16_t* __restrict__ A, const bf16_t* __restrict__ W,
    const float* __restrict__ resid, float* __restrict__ outf,
    bf16_t* __restrict__ outb, int M, int N, int K)
{
    int w = threadIdx.x >> 5, lane = threadIdx.x & 31;
    int half = (lane >> 4) & 1, ln = lane & 15;
    int m0 = blockIdx.x * 64 + (w & 3) * 16;
    int n0 = blockIdx.y * 128 + (w >> 2) * 64;

    FragC acc[4];
    v8f vz = {};
#pragma unroll
    for (int jn = 0; jn < 4; jn++) acc[jn].v = vz;

    const bf16_t* Arow  = A + (size_t)(m0 + ln) * K;
    const bf16_t* Wrow0 = W + (size_t)(n0 + ln) * K;

    for (int k0 = 0; k0 < K; k0 += 32) {
        FragAB a, b[4];
#pragma unroll
        for (int j = 0; j < 8; j++)
            a.u[j] = *(const uint32_t*)(Arow + k0 + koffA(j, half));
#pragma unroll
        for (int jn = 0; jn < 4; jn++) {
            const bf16_t* Wrow = Wrow0 + (size_t)(jn * 16) * K;
#pragma unroll
            for (int j = 0; j < 8; j++)
                b[jn].u[j] = *(const uint32_t*)(Wrow + k0 + koffB(j, half));
        }
        __builtin_prefetch(Wrow0 + k0 + 256, 0, 1);   // global_prefetch_b8
#pragma unroll
        for (int jn = 0; jn < 4; jn++)
            acc[jn].v = wmma_bf16(a.v, b[jn].v, acc[jn].v);
    }

#pragma unroll
    for (int jn = 0; jn < 4; jn++) {
#pragma unroll
        for (int r = 0; r < 8; r++) {
            int row = m0 + r + half * 8;   // C layout: lanes 0-15 -> M=r, 16-31 -> M=r+8
            int col = n0 + jn * 16 + ln;
            size_t o = (size_t)row * N + col;
            float v = acc[jn].f[r];
            if (MODE == 0)      outf[o] = v;
            else if (MODE == 1) outf[o] = resid[o] + v;
            else if (MODE == 2) { float q = v > 0.f ? v * v : 0.f; outb[o] = (bf16_t)q; }
            else                outf[o] = SOFTCAP_ * tanhf(v * (1.f / SOFTCAP_));
        }
    }
}

// ---------------------------------------------------------------------------
// RoPE + per-head RMS (*1.2) for q/k, value-embedding gate for v.
// V is written TRANSPOSED to vt[NKVH*HDIM][T_SEQ] so the flash P@V
// B-fragments become contiguous dword loads (K-consecutive key pairs).
// 10 waves per token: waves 0-5 q heads, 6-7 k heads, 8-9 v heads.
// ---------------------------------------------------------------------------
__global__ void __launch_bounds__(320) k_rope_ve(
    const float* __restrict__ qf, const float* __restrict__ kf,
    const float* __restrict__ vf, const float* __restrict__ hf,
    const int* __restrict__ idx,
    const float* __restrict__ ve_emb, const float* __restrict__ ve_gw,
    bf16_t* __restrict__ qb, bf16_t* __restrict__ kb, bf16_t* __restrict__ vt,
    int is_ve, int ve_idx)
{
    int t = blockIdx.x;
    int w = threadIdx.x >> 5, lane = threadIdx.x & 31;
    const float LOGB = 11.512925464970229f;   // ln(100000)

    if (w < 8) {
        const float* src;
        bf16_t* dst;
        if (w < 6) { src = qf + (size_t)t * CDIM  + w * HDIM;        dst = qb + (size_t)t * CDIM  + w * HDIM; }
        else       { src = kf + (size_t)t * KVDIM + (w - 6) * HDIM;  dst = kb + (size_t)t * KVDIM + (w - 6) * HDIM; }
        float o1[2], o2[2];
        float ss = 0.f;
#pragma unroll
        for (int p = 0; p < 2; p++) {
            int j = lane + p * 32;                 // rotary pair index 0..63
            float inv = __expf(-((float)(2 * j) / (float)HDIM) * LOGB);
            float ang = (float)t * inv;
            float sn, cs;
            __sincosf(ang, &sn, &cs);
            float x1 = src[j], x2 = src[j + 64];
            o1[p] =  x1 * cs + x2 * sn;
            o2[p] = -x1 * sn + x2 * cs;
            ss += o1[p] * o1[p] + o2[p] * o2[p];
        }
        for (int m = 16; m >= 1; m >>= 1) ss += __shfl_xor(ss, m, 32);
        float r = rsqrtf(ss / HDIM + 1e-6f) * 1.2f;
#pragma unroll
        for (int p = 0; p < 2; p++) {
            int j = lane + p * 32;
            dst[j]      = (bf16_t)(o1[p] * r);
            dst[j + 64] = (bf16_t)(o2[p] * r);
        }
    } else {
        int kvh = w - 8;
        const float* src = vf + (size_t)t * KVDIM + kvh * HDIM;
        float g = 0.f;
        const float* ve = nullptr;
        if (is_ve) {
            float part = (lane < 12)
                ? hf[(size_t)t * CDIM + lane] * ve_gw[((size_t)ve_idx * NKVH + kvh) * 12 + lane]
                : 0.f;
            for (int m = 16; m >= 1; m >>= 1) part += __shfl_xor(part, m, 32);
            g = 3.f / (1.f + __expf(-part));
            int id = idx[t];
            ve = ve_emb + ((size_t)ve_idx * VOCAB + id) * KVDIM + kvh * HDIM;
        }
#pragma unroll
        for (int mq = 0; mq < 4; mq++) {
            int d = lane + mq * 32;
            float vv = src[d];
            if (is_ve) vv += g * ve[d];
            // transposed store: vt[(kvh*HDIM + d)][t]
            vt[(size_t)(kvh * HDIM + d) * T_SEQ + t] = (bf16_t)vv;
        }
    }
}

// ---------------------------------------------------------------------------
// Flash attention, GQA (6 q heads / 2 kv heads), HD=128, causal + window mask.
// 8 waves per block; each wave processes 16 query rows for one head.
// QK^T and P@V both via v_wmma_f32_16x16x32_bf16; P re-fragmented through LDS.
// V comes in transposed (vt[dim][key]); fragments preloaded in groups so
// b128 loads clause and WMMAs overlap outstanding loads.
// ---------------------------------------------------------------------------
__global__ void __launch_bounds__(256) k_flash(
    const bf16_t* __restrict__ Q, const bf16_t* __restrict__ K,
    const bf16_t* __restrict__ VT, bf16_t* __restrict__ Y, int win)
{
    __shared__ bf16_t Pb[8][512];     // per-wave 16x32 probability tile
    int w = threadIdx.x >> 5, lane = threadIdx.x & 31;
    int half = (lane >> 4) & 1, ln = lane & 15;
    int q0   = blockIdx.x * 128 + w * 16;
    int head = blockIdx.y;
    int kvh  = head / (NHEAD / NKVH);

    // preload the 16x128 Q block as 4 A-fragments (K-steps of 32)
    FragAB aq[4];
    {
        const bf16_t* qp = Q + (size_t)(q0 + ln) * CDIM + head * HDIM;
#pragma unroll
        for (int kk = 0; kk < 4; kk++)
#pragma unroll
            for (int j = 0; j < 8; j++)
                aq[kk].u[j] = *(const uint32_t*)(qp + kk * 32 + koffA(j, half));
    }

    float mrow[8], lrow[8];
    FragC o[8];
    v8f vz = {};
#pragma unroll
    for (int r = 0; r < 8; r++) { mrow[r] = -3.0e38f; lrow[r] = 0.f; }
#pragma unroll
    for (int j = 0; j < 8; j++) o[j].v = vz;

    int kstart = 0;
    if (win > 0) { int lo = q0 - win; if (lo > 0) kstart = lo & ~31; }
    const float scale = 0.08838834764831845f;   // 1/sqrt(128)

    for (int kb = kstart; kb <= q0 + 15; kb += 32) {
        // S tiles: 16x32 scores as two 16x16 WMMA accumulations over HD=128;
        // the 4 K-fragments of each key tile are preloaded before the WMMAs.
        FragC st[2];
#pragma unroll
        for (int t2 = 0; t2 < 2; t2++) {
            st[t2].v = vz;
            const bf16_t* kp = K + (size_t)(kb + t2 * 16 + ln) * KVDIM + kvh * HDIM;
            FragAB bk[4];
#pragma unroll
            for (int kk = 0; kk < 4; kk++)
#pragma unroll
                for (int j = 0; j < 8; j++)
                    bk[kk].u[j] = *(const uint32_t*)(kp + kk * 32 + koffB(j, half));
#pragma unroll
            for (int kk = 0; kk < 4; kk++)
                st[t2].v = wmma_bf16(aq[kk].v, bk[kk].v, st[t2].v);
        }
        // online softmax over the 32 keys of this block
#pragma unroll
        for (int r = 0; r < 8; r++) {
            int row = q0 + r + half * 8;
            float s0 = st[0].f[r] * scale, s1 = st[1].f[r] * scale;
            int c0 = kb + ln, c1 = kb + 16 + ln;
            if (!(c0 <= row && (win < 0 || row - c0 <= win))) s0 = -3.0e38f;
            if (!(c1 <= row && (win < 0 || row - c1 <= win))) s1 = -3.0e38f;
            float mx = fmaxf(s0, s1);
            for (int m = 1; m < 16; m <<= 1) mx = fmaxf(mx, __shfl_xor(mx, m, 16));
            float mnew  = fmaxf(mrow[r], mx);
            float alpha = __expf(mrow[r] - mnew);
            float p0 = (s0 < -1e37f) ? 0.f : __expf(s0 - mnew);
            float p1 = (s1 < -1e37f) ? 0.f : __expf(s1 - mnew);
            float rs = p0 + p1;
            for (int m = 1; m < 16; m <<= 1) rs += __shfl_xor(rs, m, 16);
            lrow[r] = lrow[r] * alpha + rs;
            mrow[r] = mnew;
#pragma unroll
            for (int j = 0; j < 8; j++) o[j].f[r] *= alpha;
            int rl = r + half * 8;
            Pb[w][rl * 32 + ln]      = (bf16_t)p0;    // wave-private LDS (in-order DS)
            Pb[w][rl * 32 + 16 + ln] = (bf16_t)p1;
        }
        // re-fragment P (C layout -> A layout) through LDS, then P @ V.
        FragAB ap;
#pragma unroll
        for (int j = 0; j < 8; j++)
            ap.u[j] = *(const uint32_t*)&Pb[w][ln * 32 + koffA(j, half)];
        // V fragments in two groups of 4: clause 8 b128 loads, then 4 WMMAs.
#pragma unroll
        for (int g = 0; g < 2; g++) {
            FragAB bv[4];
#pragma unroll
            for (int j4 = 0; j4 < 4; j4++) {
                int j8 = g * 4 + j4;
                // B[k][n] = VT[(kvh*HDIM + j8*16 + n)][kb + k]; key pairs contiguous
                const bf16_t* vrow = VT + (size_t)(kvh * HDIM + j8 * 16 + ln) * T_SEQ + kb;
#pragma unroll
                for (int j = 0; j < 8; j++)
                    bv[j4].u[j] = *(const uint32_t*)(vrow + koffB(j, half));
            }
#pragma unroll
            for (int j4 = 0; j4 < 4; j4++)
                o[g * 4 + j4].v = wmma_bf16(ap.v, bv[j4].v, o[g * 4 + j4].v);
        }
    }
    // epilogue: normalize and emit bf16 activations for attn_proj GEMM
#pragma unroll
    for (int j8 = 0; j8 < 8; j8++) {
#pragma unroll
        for (int r = 0; r < 8; r++) {
            int row = q0 + r + half * 8;
            int col = head * HDIM + j8 * 16 + ln;
            float y = o[j8].f[r] / lrow[r];
            Y[(size_t)row * CDIM + col] = (bf16_t)y;
        }
    }
}

// ---------------------------------------------------------------------------
// Host orchestration
// ---------------------------------------------------------------------------
extern "C" void kernel_launch(void* const* d_in, const int* in_sizes, int n_in,
                              void* d_out, int out_size, void* d_ws, size_t ws_size,
                              hipStream_t stream)
{
    (void)in_sizes; (void)n_in; (void)out_size; (void)ws_size;
    const int*   idx   = (const int*)  d_in[0];
    const float* wte   = (const float*)d_in[1];
    const float* q_w   = (const float*)d_in[2];
    const float* k_w   = (const float*)d_in[3];
    const float* v_w   = (const float*)d_in[4];
    const float* o_w   = (const float*)d_in[5];
    const float* ve_gw = (const float*)d_in[6];
    const float* fc_w  = (const float*)d_in[7];
    const float* mp_w  = (const float*)d_in[8];
    const float* ve_e  = (const float*)d_in[9];
    const float* lm_w  = (const float*)d_in[10];
    const float* rlam  = (const float*)d_in[11];
    const float* xlam  = (const float*)d_in[12];
    const float* sgw   = (const float*)d_in[13];
    const float* slam  = (const float*)d_in[14];
    const float* blam  = (const float*)d_in[15];
    float* out = (float*)d_out;

    char* ws = (char*)d_ws;
    size_t off = 0;
    auto carve = [&](size_t bytes) -> char* {
        char* p = ws + off;
        off = (off + bytes + 255) & ~(size_t)255;
        return p;
    };
    const size_t TC = (size_t)T_SEQ * CDIM;

    bf16_t* wqb = (bf16_t*)carve((size_t)NLAY * CDIM  * CDIM * 2);
    bf16_t* wkb = (bf16_t*)carve((size_t)NLAY * KVDIM * CDIM * 2);
    bf16_t* wvb = (bf16_t*)carve((size_t)NLAY * KVDIM * CDIM * 2);
    bf16_t* wob = (bf16_t*)carve((size_t)NLAY * CDIM  * CDIM * 2);
    bf16_t* wfb = (bf16_t*)carve((size_t)NLAY * 4 * CDIM * CDIM * 2);
    bf16_t* wmb = (bf16_t*)carve((size_t)NLAY * CDIM * 4 * CDIM * 2);
    bf16_t* wlb = (bf16_t*)carve((size_t)VOCAB * CDIM * 2);
    float*  x   = (float*) carve(TC * 4);
    float*  x0  = (float*) carve(TC * 4);
    float*  xbk = (float*) carve(TC * 4);
    float*  hf  = (float*) carve(TC * 4);
    bf16_t* hb  = (bf16_t*)carve(TC * 2);
    float*  qf  = (float*) carve(TC * 4);
    float*  kf  = (float*) carve((size_t)T_SEQ * KVDIM * 4);
    float*  vf  = (float*) carve((size_t)T_SEQ * KVDIM * 4);
    bf16_t* qb  = (bf16_t*)carve(TC * 2);
    bf16_t* kb  = (bf16_t*)carve((size_t)T_SEQ * KVDIM * 2);
    bf16_t* vt  = (bf16_t*)carve((size_t)NKVH * HDIM * T_SEQ * 2);  // transposed V
    bf16_t* yb  = (bf16_t*)carve(TC * 2);
    bf16_t* mb  = (bf16_t*)carve((size_t)T_SEQ * 4 * CDIM * 2);

    auto cvt = [&](const float* s, bf16_t* d, size_t n) {
        k_cvt<<<dim3((unsigned)((n + 255) / 256)), dim3(256), 0, stream>>>(s, d, (long)n);
    };
    cvt(q_w,  wqb, (size_t)NLAY * CDIM  * CDIM);
    cvt(k_w,  wkb, (size_t)NLAY * KVDIM * CDIM);
    cvt(v_w,  wvb, (size_t)NLAY * KVDIM * CDIM);
    cvt(o_w,  wob, (size_t)NLAY * CDIM  * CDIM);
    cvt(fc_w, wfb, (size_t)NLAY * 4 * CDIM * CDIM);
    cvt(mp_w, wmb, (size_t)NLAY * CDIM * 4 * CDIM);
    cvt(lm_w, wlb, (size_t)VOCAB * CDIM);

    k_embed<<<dim3(T_SEQ), dim3(256), 0, stream>>>(idx, wte, sgw, slam, x, x0);

    dim3 blk(256);
    for (int i = 0; i < NLAY; i++) {
        // x = rl*x + xl*x0 ; h = rms(x)
        k_norm<<<T_SEQ, 256, 0, stream>>>(x, x0, rlam, i, xlam, i, 1.f, hf, hb);
        // Q/K/V projections
        k_gemm<0><<<dim3(T_SEQ / 64, CDIM  / 128), blk, 0, stream>>>(
            hb, wqb + (size_t)i * CDIM  * CDIM, nullptr, qf, nullptr, T_SEQ, CDIM,  CDIM);
        k_gemm<0><<<dim3(T_SEQ / 64, KVDIM / 128), blk, 0, stream>>>(
            hb, wkb + (size_t)i * KVDIM * CDIM, nullptr, kf, nullptr, T_SEQ, KVDIM, CDIM);
        k_gemm<0><<<dim3(T_SEQ / 64, KVDIM / 128), blk, 0, stream>>>(
            hb, wvb + (size_t)i * KVDIM * CDIM, nullptr, vf, nullptr, T_SEQ, KVDIM, CDIM);
        // rope + head rms + value embedding gate (+ V transpose)
        int is_ve = (i % 2) == 1;
        k_rope_ve<<<T_SEQ, 320, 0, stream>>>(qf, kf, vf, hf, idx, ve_e, ve_gw,
                                             qb, kb, vt, is_ve, i / 2);
        // flash attention
        int win = ((i % 4) == 3) ? -1 : WINSZ;
        k_flash<<<dim3(T_SEQ / 128, NHEAD), blk, 0, stream>>>(qb, kb, vt, yb, win);
        // attn proj with fused residual add (in place on x)
        k_gemm<1><<<dim3(T_SEQ / 64, CDIM / 128), blk, 0, stream>>>(
            yb, wob + (size_t)i * CDIM * CDIM, x, x, nullptr, T_SEQ, CDIM, CDIM);
        if (i == NLAY / 2 - 1)
            hipMemcpyAsync(xbk, x, TC * 4, hipMemcpyDeviceToDevice, stream);
        // h2 = rms(x)
        k_norm<<<T_SEQ, 256, 0, stream>>>(x, x, nullptr, 0, nullptr, 0, 0.f, hf, hb);
        // MLP: fc (relu^2 -> bf16), then proj with fused residual
        k_gemm<2><<<dim3(T_SEQ / 64, 4 * CDIM / 128), blk, 0, stream>>>(
            hb, wfb + (size_t)i * 4 * CDIM * CDIM, nullptr, nullptr, mb, T_SEQ, 4 * CDIM, CDIM);
        k_gemm<1><<<dim3(T_SEQ / 64, CDIM / 128), blk, 0, stream>>>(
            mb, wmb + (size_t)i * CDIM * 4 * CDIM, x, x, nullptr, T_SEQ, CDIM, 4 * CDIM);
    }
    // x = x - backout_lambda * x_backout ; h = rms(x)
    k_norm<<<T_SEQ, 256, 0, stream>>>(x, xbk, nullptr, 0, blam, 0, -1.f, hf, hb);
    // lm head with fused softcap tanh -> d_out (f32)
    k_gemm<3><<<dim3(T_SEQ / 64, VOCAB / 128), blk, 0, stream>>>(
        hb, wlb, nullptr, out, nullptr, T_SEQ, VOCAB, CDIM);
}
